// MultiHeadAttention_23648089931933
// MI455X (gfx1250) — compile-verified, matrix operand order checked by reading, not run on hardware
//
#include <hip/hip_runtime.h>
#include <hip/hip_bf16.h>

// MI455X / gfx1250: wave32, WMMA 16x16x32 bf16, async global->LDS, 320KB LDS/WGP.

typedef __attribute__((ext_vector_type(16))) __bf16 v16bf;
typedef __attribute__((ext_vector_type(8)))  __bf16 v8bf;
typedef __attribute__((ext_vector_type(8)))  float  v8f;
typedef int v4i __attribute__((vector_size(16)));
typedef __attribute__((address_space(1))) v4i* gptr4;
typedef __attribute__((address_space(3))) v4i* lptr4;

#define H_DIM  2048
#define NHEADS 16
#define DHEAD  128
#define SEQ    2048
#define BATCH  4
#define ROWS   (BATCH * SEQ)          // 8192
#define N_QKV  (3 * H_DIM)            // 6144

#if __has_builtin(__builtin_amdgcn_global_load_async_to_lds_b128)
#define ASYNC_LDS 1
#else
#define ASYNC_LDS 0
#endif

__device__ __forceinline__ void cp16(const __bf16* g, __bf16* l) {
#if ASYNC_LDS
    __builtin_amdgcn_global_load_async_to_lds_b128((gptr4)(g), (lptr4)(l), 0, 0);
#else
    *(v8bf*)l = *(const v8bf*)g;
#endif
}

__device__ __forceinline__ void async_wait_le4() {
#if ASYNC_LDS
#if __has_builtin(__builtin_amdgcn_s_wait_asynccnt)
    __builtin_amdgcn_s_wait_asynccnt(4);
#else
    asm volatile("s_wait_asynccnt 0x4" ::: "memory");
#endif
#endif
}
__device__ __forceinline__ void async_wait_le8() {
#if ASYNC_LDS
#if __has_builtin(__builtin_amdgcn_s_wait_asynccnt)
    __builtin_amdgcn_s_wait_asynccnt(8);
#else
    asm volatile("s_wait_asynccnt 0x8" ::: "memory");
#endif
#endif
}
__device__ __forceinline__ void async_wait_0() {
#if ASYNC_LDS
#if __has_builtin(__builtin_amdgcn_s_wait_asynccnt)
    __builtin_amdgcn_s_wait_asynccnt(0);
#else
    asm volatile("s_wait_asynccnt 0x0" ::: "memory");
#endif
#endif
}

__device__ __forceinline__ __bf16 f2bf(float f) {
    unsigned u = __builtin_bit_cast(unsigned, f);
    unsigned r = (u + 0x7FFFu + ((u >> 16) & 1u)) >> 16;
    return __builtin_bit_cast(__bf16, (unsigned short)r);
}

__device__ __forceinline__ v16bf combine(v8bf lo, v8bf hi) {
    return __builtin_shufflevector(lo, hi, 0,1,2,3,4,5,6,7,8,9,10,11,12,13,14,15);
}

__device__ __forceinline__ v8f v8f_zero() {
    v8f z;
#pragma unroll
    for (int i = 0; i < 8; ++i) z[i] = 0.0f;
    return z;
}

__device__ __forceinline__ float blockReduceSum(float v) {
    __shared__ float smem[8];
    for (int off = 16; off > 0; off >>= 1) v += __shfl_xor(v, off, 32);
    int lane = threadIdx.x & 31, wid = threadIdx.x >> 5;
    if (lane == 0) smem[wid] = v;
    __syncthreads();
    v = (lane < 8) ? smem[lane] : 0.0f;
    if (wid == 0) {
        for (int off = 4; off > 0; off >>= 1) v += __shfl_xor(v, off, 32);
        if (lane == 0) smem[0] = v;
    }
    __syncthreads();
    v = smem[0];
    __syncthreads();
    return v;
}

// ---------------- weight convert + transpose: W[K][N] f32 -> WT[N][K] bf16 ----
__global__ void k_transpose_bf16(const float* __restrict__ W, __bf16* __restrict__ WT,
                                 int K, int N) {
    int n = blockIdx.x * 16 + threadIdx.x;
    int k = blockIdx.y * 16 + threadIdx.y;
    WT[(size_t)n * K + k] = f2bf(W[(size_t)k * N + n]);
}

// ---------------- LayerNorm(x) -> bf16 ---------------------------------------
__global__ __launch_bounds__(256) void k_ln_x(const float* __restrict__ x,
                                              const float* __restrict__ w,
                                              const float* __restrict__ b,
                                              __bf16* __restrict__ xn) {
    const int row = blockIdx.x;
    const float* xr = x + (size_t)row * H_DIM;
    float v[8], s = 0.0f;
#pragma unroll
    for (int j = 0; j < 8; ++j) { v[j] = xr[threadIdx.x + j * 256]; s += v[j]; }
    s = blockReduceSum(s);
    const float mu = s * (1.0f / H_DIM);
    float s2 = 0.0f;
#pragma unroll
    for (int j = 0; j < 8; ++j) { float d = v[j] - mu; s2 += d * d; }
    s2 = blockReduceSum(s2);
    const float rs = rsqrtf(s2 * (1.0f / H_DIM) + 1e-5f);
#pragma unroll
    for (int j = 0; j < 8; ++j) {
        int idx = threadIdx.x + j * 256;
        xn[(size_t)row * H_DIM + idx] = f2bf((v[j] - mu) * rs * w[idx] + b[idx]);
    }
}

// ---------------- bf16 WMMA GEMM: C[M][N] = A[M][K] @ Bt[N][K]^T + bias ------
// Block tile 128x128, 8 waves (2x4), each wave 64x32 = 4x2 WMMA subtiles.
// Double-buffered LDS fed by async global->LDS copies (ASYNCcnt pipelined).
__global__ __launch_bounds__(256) void k_gemm_bf16(const __bf16* __restrict__ A,
                                                   const __bf16* __restrict__ Bt,
                                                   const float* __restrict__ bias,
                                                   float* __restrict__ C,
                                                   int M, int N, int K) {
    __shared__ __align__(16) __bf16 As[2][128][40];
    __shared__ __align__(16) __bf16 Bs[2][128][40];
    const int t     = threadIdx.x;
    const int lane  = t & 31;
    const int wave  = t >> 5;
    const int waveM = wave >> 2;   // 0..1
    const int waveN = wave & 3;    // 0..3
    const int Mbase = blockIdx.y * 128;
    const int Nbase = blockIdx.x * 128;

    v8f acc[4][2];
#pragma unroll
    for (int mi = 0; mi < 4; ++mi)
#pragma unroll
        for (int ni = 0; ni < 2; ++ni) acc[mi][ni] = v8f_zero();

    const int ldRow = t >> 1;
    const int ldCol = (t & 1) * 16;
    const int khalf = (lane < 16) ? 0 : 8;
    const int lr    = lane & 15;

    auto issue = [&](int k0, int bsel) {
        const __bf16* ag = A  + (size_t)(Mbase + ldRow) * K + k0 + ldCol;
        const __bf16* bg = Bt + (size_t)(Nbase + ldRow) * K + k0 + ldCol;
        cp16(ag,     &As[bsel][ldRow][ldCol]);
        cp16(ag + 8, &As[bsel][ldRow][ldCol + 8]);
        cp16(bg,     &Bs[bsel][ldRow][ldCol]);
        cp16(bg + 8, &Bs[bsel][ldRow][ldCol + 8]);
    };

    issue(0, 0);
    int buf = 0;
    for (int k0 = 0; k0 < K; k0 += 32, buf ^= 1) {
        if (k0 + 32 < K) { issue(k0 + 32, buf ^ 1); async_wait_le4(); }
        else             { async_wait_0(); }
        __syncthreads();

        v16bf a[4], b[2];
#pragma unroll
        for (int mi = 0; mi < 4; ++mi) {
            const __bf16* p = &As[buf][waveM * 64 + mi * 16 + lr][khalf];
            a[mi] = combine(*(const v8bf*)p, *(const v8bf*)(p + 16));
        }
#pragma unroll
        for (int ni = 0; ni < 2; ++ni) {
            const __bf16* p = &Bs[buf][waveN * 32 + ni * 16 + lr][khalf];
            b[ni] = combine(*(const v8bf*)p, *(const v8bf*)(p + 16));
        }
#pragma unroll
        for (int mi = 0; mi < 4; ++mi)
#pragma unroll
            for (int ni = 0; ni < 2; ++ni)
                acc[mi][ni] = __builtin_amdgcn_wmma_f32_16x16x32_bf16(
                    false, a[mi], false, b[ni], (short)0, acc[mi][ni], false, false);
        __syncthreads();
    }

    const int rbase = Mbase + waveM * 64 + ((lane >= 16) ? 8 : 0);
    const int cbase = Nbase + waveN * 32 + lr;
#pragma unroll
    for (int mi = 0; mi < 4; ++mi)
#pragma unroll
        for (int ni = 0; ni < 2; ++ni) {
            int col = cbase + ni * 16;
            float bv = bias ? bias[col] : 0.0f;
#pragma unroll
            for (int i = 0; i < 8; ++i)
                C[(size_t)(rbase + mi * 16 + i) * N + col] = acc[mi][ni][i] + bv;
        }
}

// ---------------- Q/K LayerNorm + RoPE + head reshape ------------------------
// qkv f32 [8192][6144] -> qT,kT bf16 [b][h][L][128], vT bf16 [b][h][128][L]
__global__ __launch_bounds__(256) void k_qk_rope(const float* __restrict__ qkv,
                                                 const float* __restrict__ qw,
                                                 const float* __restrict__ kw,
                                                 __bf16* __restrict__ qT,
                                                 __bf16* __restrict__ kT,
                                                 __bf16* __restrict__ vT) {
    const int row = blockIdx.x;
    const int b   = row >> 11;
    const int l   = row & (SEQ - 1);
    const float* qr = qkv + (size_t)row * N_QKV;
    const float* kr = qr + H_DIM;
    const float* vr = qr + 2 * H_DIM;

    float qv[8], kv[8], sq = 0.0f, sk = 0.0f;
#pragma unroll
    for (int j = 0; j < 8; ++j) {
        int idx = threadIdx.x + j * 256;
        qv[j] = qr[idx]; kv[j] = kr[idx]; sq += qv[j]; sk += kv[j];
    }
    sq = blockReduceSum(sq);
    sk = blockReduceSum(sk);
    const float qmu = sq * (1.0f / H_DIM), kmu = sk * (1.0f / H_DIM);
    float q2 = 0.0f, k2 = 0.0f;
#pragma unroll
    for (int j = 0; j < 8; ++j) {
        float d = qv[j] - qmu; q2 += d * d;
        d = kv[j] - kmu;       k2 += d * d;
    }
    q2 = blockReduceSum(q2);
    k2 = blockReduceSum(k2);
    const float qrs = rsqrtf(q2 * (1.0f / H_DIM) + 1e-5f);
    const float krs = rsqrtf(k2 * (1.0f / H_DIM) + 1e-5f);

#pragma unroll
    for (int j = 0; j < 8; ++j) {
        int idx = threadIdx.x + j * 256;
        int h = idx >> 7, d = idx & 127, jj = d & 63;
        float f  = (float)l * __expf(-(float)(2 * jj) * (9.2103403719761836f / 128.0f));
        float cs = __cosf(f), sn = __sinf(f);
        int partner = (d < 64) ? idx + 64 : idx - 64;
        float qn = (qv[j] - qmu) * qrs * qw[idx];
        float qp = (qr[partner] - qmu) * qrs * qw[partner];
        float kn = (kv[j] - kmu) * krs * kw[idx];
        float kp = (kr[partner] - kmu) * krs * kw[partner];
        float qo = (d < 64) ? (qn * cs - qp * sn) : (qn * cs + qp * sn);
        float ko = (d < 64) ? (kn * cs - kp * sn) : (kn * cs + kp * sn);
        size_t qi = ((size_t)(b * NHEADS + h) * SEQ + l) * DHEAD + d;
        qT[qi] = f2bf(qo);
        kT[qi] = f2bf(ko);
        vT[((size_t)(b * NHEADS + h) * DHEAD + d) * SEQ + l] = f2bf(vr[idx]);
    }
}

// ---------------- Flash attention: WMMA scores + WMMA P@V --------------------
// grid (L/128, NH, B); 8 waves/block, each wave owns 16 q rows.
// K (64x128) and V (128x64, d-major) staged in double-buffered LDS via async
// copies, shared by all 8 waves (8x global-traffic reduction vs per-wave loads).
__global__ __launch_bounds__(256) void k_attn(const __bf16* __restrict__ qT,
                                              const __bf16* __restrict__ kT,
                                              const __bf16* __restrict__ vT,
                                              __bf16* __restrict__ aOut) {
    __shared__ __align__(16) __bf16 Ks[2][64][136];
    __shared__ __align__(16) __bf16 Vs[2][128][72];
    __shared__ __align__(16) __bf16 Pbuf[8][16][64];
    const int t     = threadIdx.x;
    const int lane  = t & 31;
    const int wave  = t >> 5;
    const int h     = blockIdx.y, b = blockIdx.z;
    const int q0    = blockIdx.x * 128 + wave * 16;
    const size_t bh = (size_t)(b * NHEADS + h);
    const __bf16* qBase = qT + bh * SEQ * DHEAD;
    const __bf16* kBase = kT + bh * SEQ * DHEAD;
    const __bf16* vBase = vT + bh * DHEAD * SEQ;

    const int lr    = lane & 15;
    const int khalf = (lane < 16) ? 0 : 8;
    const float scale = 0.08838834764831845f;   // 1/sqrt(128)

    v16bf qf[4];
    {
        const __bf16* qp = qBase + (size_t)(q0 + lr) * DHEAD;
#pragma unroll
        for (int kc = 0; kc < 4; ++kc) {
            const __bf16* p = qp + kc * 32 + khalf;
            qf[kc] = combine(*(const v8bf*)p, *(const v8bf*)(p + 16));
        }
    }

    auto issueKV = [&](int kb, int bsel) {
        {   // K tile: 64 key rows x 128 d, 32 bf16 per thread
            int r = t >> 2, c = (t & 3) * 32;
            const __bf16* g = kBase + (size_t)(kb + r) * DHEAD + c;
            __bf16* l = &Ks[bsel][r][c];
            cp16(g, l); cp16(g + 8, l + 8); cp16(g + 16, l + 16); cp16(g + 24, l + 24);
        }
        {   // V tile: 128 d rows x 64 keys, 32 bf16 per thread
            int r = t >> 1, c = (t & 1) * 32;
            const __bf16* g = vBase + (size_t)r * SEQ + kb + c;
            __bf16* l = &Vs[bsel][r][c];
            cp16(g, l); cp16(g + 8, l + 8); cp16(g + 16, l + 16); cp16(g + 24, l + 24);
        }
    };

    v8f o[8];
    float m[8], ssum[8];
#pragma unroll
    for (int i = 0; i < 8; ++i) { o[i] = v8f_zero(); m[i] = -3.0e38f; ssum[i] = 0.0f; }
    __bf16* pb = &Pbuf[wave][0][0];

    issueKV(0, 0);
    int buf = 0;
    for (int kb = 0; kb < SEQ; kb += 64, buf ^= 1) {
        if (kb + 64 < SEQ) { issueKV(kb + 64, buf ^ 1); async_wait_le8(); }
        else               { async_wait_0(); }
        __syncthreads();

        // S(16x64) = Q @ K^T over D=128 (4 chunks of 32)
        v8f s[4];
#pragma unroll
        for (int ni = 0; ni < 4; ++ni) {
            s[ni] = v8f_zero();
#pragma unroll
            for (int kc = 0; kc < 4; ++kc) {
                const __bf16* p = &Ks[buf][ni * 16 + lr][kc * 32 + khalf];
                v16bf kf = combine(*(const v8bf*)p, *(const v8bf*)(p + 16));
                s[ni] = __builtin_amdgcn_wmma_f32_16x16x32_bf16(
                    false, qf[kc], false, kf, (short)0, s[ni], false, false);
            }
        }
        // online softmax per row (8 rows per lane-half)
        float corr[8];
#pragma unroll
        for (int i = 0; i < 8; ++i) {
            float mx = fmaxf(fmaxf(s[0][i], s[1][i]), fmaxf(s[2][i], s[3][i])) * scale;
            for (int off = 1; off < 16; off <<= 1) mx = fmaxf(mx, __shfl_xor(mx, off, 32));
            float mnew = fmaxf(m[i], mx);
            corr[i] = __expf(m[i] - mnew);
            float rsum = 0.0f;
#pragma unroll
            for (int ni = 0; ni < 4; ++ni) {
                float p = __expf(s[ni][i] * scale - mnew);
                s[ni][i] = p;
                rsum += p;
            }
            for (int off = 1; off < 16; off <<= 1) rsum += __shfl_xor(rsum, off, 32);
            ssum[i] = ssum[i] * corr[i] + rsum;
            m[i] = mnew;
        }
#pragma unroll
        for (int nd = 0; nd < 8; ++nd)
#pragma unroll
            for (int i = 0; i < 8; ++i) o[nd][i] *= corr[i];

        // redistribute P through LDS (C-layout -> A-layout), bf16
        const int rofs = (lane >= 16) ? 8 : 0;
#pragma unroll
        for (int ni = 0; ni < 4; ++ni)
#pragma unroll
            for (int i = 0; i < 8; ++i)
                pb[(i + rofs) * 64 + ni * 16 + lr] = f2bf(s[ni][i]);
        __syncthreads();
        v16bf pf[2];
#pragma unroll
        for (int kc = 0; kc < 2; ++kc) {
            const __bf16* p = pb + (size_t)lr * 64 + kc * 32 + khalf;
            pf[kc] = combine(*(const v8bf*)p, *(const v8bf*)(p + 16));
        }
        // O(16x128) += P(16x64) @ V(64x128), V staged as Vs[d][key]
#pragma unroll
        for (int nd = 0; nd < 8; ++nd)
#pragma unroll
            for (int kc = 0; kc < 2; ++kc) {
                const __bf16* vp = &Vs[buf][nd * 16 + lr][kc * 32 + khalf];
                v16bf vf = combine(*(const v8bf*)vp, *(const v8bf*)(vp + 16));
                o[nd] = __builtin_amdgcn_wmma_f32_16x16x32_bf16(
                    false, pf[kc], false, vf, (short)0, o[nd], false, false);
            }
        __syncthreads();
    }

    float inv[8];
#pragma unroll
    for (int i = 0; i < 8; ++i) inv[i] = 1.0f / ssum[i];
    const int rowbase = q0 + ((lane >= 16) ? 8 : 0);
#pragma unroll
    for (int nd = 0; nd < 8; ++nd)
#pragma unroll
        for (int i = 0; i < 8; ++i) {
            int row = rowbase + i;
            int d   = nd * 16 + lr;
            aOut[((size_t)(b * SEQ + row)) * H_DIM + h * DHEAD + d] = f2bf(o[nd][i] * inv[i]);
        }
}

// ---------------- launch -----------------------------------------------------
extern "C" void kernel_launch(void* const* d_in, const int* in_sizes, int n_in,
                              void* d_out, int out_size, void* d_ws, size_t ws_size,
                              hipStream_t stream) {
    const float* x     = (const float*)d_in[0];
    const float* ln1_w = (const float*)d_in[1];
    const float* ln1_b = (const float*)d_in[2];
    const float* Wqkv  = (const float*)d_in[3];
    const float* bqkv  = (const float*)d_in[4];
    const float* qln_w = (const float*)d_in[5];
    const float* kln_w = (const float*)d_in[6];
    const float* Wo    = (const float*)d_in[7];
    const float* bo    = (const float*)d_in[8];
    float* out = (float*)d_out;

    char* ws = (char*)d_ws;
    __bf16* xn    = (__bf16*)(ws);                          //  32 MiB
    __bf16* WqkvT = (__bf16*)(ws + ((size_t)32  << 20));    //  24 MiB
    __bf16* WoT   = (__bf16*)(ws + ((size_t)56  << 20));    //   8 MiB
    float*  qkv   = (float*) (ws + ((size_t)64  << 20));    // 192 MiB
    __bf16* qTp   = (__bf16*)(ws + ((size_t)256 << 20));    //  32 MiB
    __bf16* kTp   = (__bf16*)(ws + ((size_t)288 << 20));    //  32 MiB
    __bf16* vTp   = (__bf16*)(ws + ((size_t)320 << 20));    //  32 MiB
    __bf16* aO    = (__bf16*)(ws + ((size_t)352 << 20));    //  32 MiB

    k_transpose_bf16<<<dim3(N_QKV / 16, H_DIM / 16), dim3(16, 16), 0, stream>>>(Wqkv, WqkvT, H_DIM, N_QKV);
    k_transpose_bf16<<<dim3(H_DIM / 16, H_DIM / 16), dim3(16, 16), 0, stream>>>(Wo, WoT, H_DIM, H_DIM);
    k_ln_x<<<ROWS, 256, 0, stream>>>(x, ln1_w, ln1_b, xn);
    k_gemm_bf16<<<dim3(N_QKV / 128, ROWS / 128), 256, 0, stream>>>(xn, WqkvT, bqkv, qkv, ROWS, N_QKV, H_DIM);
    k_qk_rope<<<ROWS, 256, 0, stream>>>(qkv, qln_w, kln_w, qTp, kTp, vTp);
    k_attn<<<dim3(SEQ / 128, NHEADS, BATCH), 256, 0, stream>>>(qTp, kTp, vTp, aO);
    k_gemm_bf16<<<dim3(H_DIM / 128, ROWS / 128), 256, 0, stream>>>(aO, WoT, bo, out, ROWS, H_DIM, H_DIM);
}
